// graphormer_79740362817882
// MI455X (gfx1250) — matrix operand order repaired and checked by reading, not compiled, not run on hardware
//
#include <hip/hip_runtime.h>

// ---------------------------------------------------------------------------
// CDNA5 (gfx1250) implementation. Heavy math via v_wmma_f32_16x16x32_bf16 with
// double-buffered async global->LDS staging (global_load_async_to_lds_b128 /
// ASYNCcnt). GEMM convention: C[M,N] = A[M,K] (bf16 row-major) x Bt[N,K]
// (bf16, B pre-transposed) + bias, optional fused GELU, optional bf16 output.
// Wave32, 8 waves / block, 128x128x32 tiles.
// ---------------------------------------------------------------------------

typedef __bf16 bf16_t;
typedef __attribute__((ext_vector_type(16))) __bf16 bf16x16;
typedef __attribute__((ext_vector_type(8)))  float  v8f;
typedef __attribute__((ext_vector_type(4)))  int    v4i;

typedef __attribute__((address_space(1))) v4i g_v4i;
typedef __attribute__((address_space(3))) v4i l_v4i;

union FragU { uint4 u[2]; bf16x16 v; };

#define TB 256

#if __has_builtin(__builtin_amdgcn_global_load_async_to_lds_b128)
#define HAS_ASYNC_LDS 1
#else
#define HAS_ASYNC_LDS 0
#endif

__device__ __forceinline__ void copy16_to_lds(const bf16_t* g, bf16_t* l)
{
#if HAS_ASYNC_LDS
    // gfx1250 async DMA: global -> LDS, tracked by ASYNCcnt (no VGPR round trip)
    __builtin_amdgcn_global_load_async_to_lds_b128(
        (g_v4i*)(uintptr_t)g, (l_v4i*)(uintptr_t)l, 0, 0);
#else
    *(uint4*)l = *(const uint4*)g;
#endif
}

__device__ __forceinline__ void wait_async4()
{
#if HAS_ASYNC_LDS
 #if __has_builtin(__builtin_amdgcn_s_wait_asynccnt)
    __builtin_amdgcn_s_wait_asynccnt(4);
 #else
    asm volatile("s_wait_asynccnt 0x4" ::: "memory");
 #endif
#endif
}

__device__ __forceinline__ void wait_async0()
{
#if HAS_ASYNC_LDS
 #if __has_builtin(__builtin_amdgcn_s_wait_asynccnt)
    __builtin_amdgcn_s_wait_asynccnt(0);
 #else
    asm volatile("s_wait_asynccnt 0x0" ::: "memory");
 #endif
#endif
}

__device__ __forceinline__ float gelu_f(float v)
{
    return 0.5f * v * (1.0f + erff(v * 0.70710678118654752f));
}

__global__ __launch_bounds__(TB)
void gemm_bf16_kernel(const bf16_t* __restrict__ A, const bf16_t* __restrict__ Bt,
                      float* __restrict__ C, bf16_t* __restrict__ Cbf,
                      const float* __restrict__ bias, float bscale, int act,
                      int M, int N, int K)
{
    // M % 128 == 0, N % 128 == 0, K % 32 == 0 (guaranteed by caller)
    __shared__ bf16_t sA[2][128][40];   // pad 8 elems (16B) keeps 16B alignment
    __shared__ bf16_t sB[2][128][40];

    const int tid  = threadIdx.x;
    const int lane = tid & 31;
    const int wave = tid >> 5;       // 0..7
    const int wm   = wave & 3;       // 4 waves along M (32 rows each)
    const int wn   = wave >> 2;      // 2 waves along N (64 cols each)
    const int bM   = blockIdx.y * 128;
    const int bN   = blockIdx.x * 128;

    v8f acc[2][4];
    const v8f vz = {0.f,0.f,0.f,0.f,0.f,0.f,0.f,0.f};
#pragma unroll
    for (int i = 0; i < 2; ++i)
#pragma unroll
        for (int j = 0; j < 4; ++j) acc[i][j] = vz;

    const int lr = tid >> 2;          // 0..63
    const int lc = (tid & 3) << 3;    // 0,8,16,24  (8 bf16 = 16B per copy)
    const int mr = lane & 15;
    const int g  = lane >> 4;

    const bf16_t* Ar0 = A  + (size_t)(bM + lr)      * K + lc;
    const bf16_t* Ar1 = A  + (size_t)(bM + lr + 64) * K + lc;
    const bf16_t* Br0 = Bt + (size_t)(bN + lr)      * K + lc;
    const bf16_t* Br1 = Bt + (size_t)(bN + lr + 64) * K + lc;

    auto stage = [&](int k0, int buf) {
        copy16_to_lds(Ar0 + k0, &sA[buf][lr][lc]);
        copy16_to_lds(Ar1 + k0, &sA[buf][lr + 64][lc]);
        copy16_to_lds(Br0 + k0, &sB[buf][lr][lc]);
        copy16_to_lds(Br1 + k0, &sB[buf][lr + 64][lc]);
    };

    // --- fragments per ISA 16-bit layouts ----------------------------------
    // A (16x32): lane l: m=l&15, g=l>>4; halves 0..7 = A[m][g*8..g*8+7],
    //            halves 8..15 = A[m][16+g*8 ..]
    // B (32x16) from Bt[N][K]: lane l: n=l&15, g=l>>4;
    //            halves 0..7 = Bt[n][g*8..g*8+7], halves 8..15 = Bt[n][16+g*8 ..]
    auto compute = [&](int buf) {
        FragU fa[2], fb[4];
#pragma unroll
        for (int tm = 0; tm < 2; ++tm) {
            int r = wm * 32 + tm * 16 + mr;
            fa[tm].u[0] = *(const uint4*)&sA[buf][r][g * 8];
            fa[tm].u[1] = *(const uint4*)&sA[buf][r][16 + g * 8];
        }
#pragma unroll
        for (int tn = 0; tn < 4; ++tn) {
            int r = wn * 64 + tn * 16 + mr;
            fb[tn].u[0] = *(const uint4*)&sB[buf][r][g * 8];
            fb[tn].u[1] = *(const uint4*)&sB[buf][r][16 + g * 8];
        }
#pragma unroll
        for (int tm = 0; tm < 2; ++tm)
#pragma unroll
            for (int tn = 0; tn < 4; ++tn)
                acc[tm][tn] = __builtin_amdgcn_wmma_f32_16x16x32_bf16(
                    false, fa[tm].v, false, fb[tn].v,
                    (short)0, acc[tm][tn], false, false);
    };

    stage(0, 0);                      // prologue: tile 0 in flight
    int cur = 0;
    int k0  = 0;

    // steady state: branch-free pipeline (issue next, wait current, compute)
    for (; k0 + 32 < K; k0 += 32) {
        stage(k0 + 32, cur ^ 1);      // next tile's 4 async copies
        wait_async4();                // first group (4) complete
        __syncthreads();              // tile `cur` visible to block
        __builtin_prefetch(Ar0 + k0 + 64, 0, 1);   // warm GL2 two tiles ahead
        __builtin_prefetch(Br0 + k0 + 64, 0, 1);   // (speculative; may drop)
        compute(cur);
        __syncthreads();              // all waves done reading `cur` before
        cur ^= 1;                     // it is re-staged next iteration
    }
    // tail: last tile
    wait_async0();
    __syncthreads();
    compute(cur);

    // C/D layout: VGPR i, lanes 0-15 -> M=i, lanes 16-31 -> M=i+8, N=lane&15
    const int cn = lane & 15;
#pragma unroll
    for (int tm = 0; tm < 2; ++tm) {
        int row0 = bM + wm * 32 + tm * 16 + g * 8;
#pragma unroll
        for (int tn = 0; tn < 4; ++tn) {
            int col = bN + wn * 64 + tn * 16 + cn;
            float bv = bias ? bias[col] * bscale : 0.0f;
#pragma unroll
            for (int i = 0; i < 8; ++i) {
                float val = acc[tm][tn][i] + bv;
                if (act) val = gelu_f(val);
                size_t idx = (size_t)(row0 + i) * N + col;
                if (C)   C[idx]   = val;
                if (Cbf) Cbf[idx] = (bf16_t)val;
            }
        }
    }
}

// --------------------------- support kernels -------------------------------

__global__ void cvt_kernel(const float* __restrict__ src, int ld,
                           bf16_t* __restrict__ dst, int rows, int cols)
{
    int i = blockIdx.x * TB + threadIdx.x;
    int n = rows * cols;
    if (i >= n) return;
    int r = i / cols, c = i - r * cols;
    dst[i] = (bf16_t)src[(size_t)r * ld + c];
}

// f32 [R,C] row-major -> bf16 [C,R] row-major (transpose + convert)
__global__ void tconv_kernel(const float* __restrict__ src,
                             bf16_t* __restrict__ dst, int R, int C)
{
    __shared__ float tile[32][33];
    int c0 = blockIdx.x * 32, r0 = blockIdx.y * 32;
    for (int i = threadIdx.y; i < 32; i += 8)
        tile[i][threadIdx.x] = src[(size_t)(r0 + i) * C + c0 + threadIdx.x];
    __syncthreads();
    for (int i = threadIdx.y; i < 32; i += 8)
        dst[(size_t)(c0 + i) * R + r0 + threadIdx.x] = (bf16_t)tile[threadIdx.x][i];
}

// BatchNorm over rows (training batch stats). Block = 32 cols x 8 rows so all
// global accesses are 128B-coalesced per wave.
__global__ __launch_bounds__(TB)
void bn_kernel(const float* __restrict__ z, int ldz, float* __restrict__ out, int ldo,
               const float* __restrict__ g, const float* __restrict__ b,
               int rows, int gmod)
{
    const int c  = blockIdx.x * 32 + threadIdx.x;
    const int ty = threadIdx.y;   // 0..7
    float s = 0.f, s2 = 0.f;
    for (int r = ty; r < rows; r += 8) {
        float v = z[(size_t)r * ldz + c];
        s += v; s2 += v * v;
    }
    __shared__ float sh1[8][32], sh2[8][32];
    sh1[ty][threadIdx.x] = s; sh2[ty][threadIdx.x] = s2;
    __syncthreads();
    if (ty == 0) {
#pragma unroll
        for (int i = 1; i < 8; ++i) { s += sh1[i][threadIdx.x]; s2 += sh2[i][threadIdx.x]; }
        float mean  = s / rows;
        float var   = s2 / rows - mean * mean;
        float scale = g[c % gmod] * rsqrtf(var + 1e-5f);
        sh1[0][threadIdx.x] = scale;
        sh2[0][threadIdx.x] = b[c % gmod] - mean * scale;
    }
    __syncthreads();
    float scale = sh1[0][threadIdx.x];
    float shift = sh2[0][threadIdx.x];
    for (int r = ty; r < rows; r += 8)
        out[(size_t)r * ldo + c] = z[(size_t)r * ldz + c] * scale + shift;
}

__global__ __launch_bounds__(TB)
void ln_kernel(const float* __restrict__ z, float* __restrict__ out,
               const float* __restrict__ g, const float* __restrict__ b, int cols)
{
    const int r = blockIdx.x;
    const float* zr = z + (size_t)r * cols;
    float s = 0.f, s2 = 0.f;
    for (int c = threadIdx.x; c < cols; c += TB) { float v = zr[c]; s += v; s2 += v * v; }
    __shared__ float sh1[TB], sh2[TB];
    sh1[threadIdx.x] = s; sh2[threadIdx.x] = s2;
    __syncthreads();
    for (int o = TB / 2; o > 0; o >>= 1) {
        if (threadIdx.x < o) { sh1[threadIdx.x] += sh1[threadIdx.x + o];
                               sh2[threadIdx.x] += sh2[threadIdx.x + o]; }
        __syncthreads();
    }
    float mean = sh1[0] / cols;
    float inv  = rsqrtf(sh2[0] / cols - mean * mean + 1e-5f);
    for (int c = threadIdx.x; c < cols; c += TB)
        out[(size_t)r * cols + c] = (zr[c] - mean) * inv * g[c] + b[c];
}

// h = (qn+kn+vn)/6 + y/2 ; qn/kn/vn are cols 0/768/1536 of [4096,2304]
__global__ void comb1_kernel(const float* __restrict__ qkvn,
                             const float* __restrict__ y,
                             float* __restrict__ h, int n)
{
    int i = blockIdx.x * TB + threadIdx.x;
    if (i >= n) return;
    int r = i / 768, c = i - r * 768;
    const float* base = qkvn + (size_t)r * 2304;
    h[i] = (base[c] + base[c + 768] + base[c + 1536]) * (1.0f / 6.0f) + 0.5f * y[i];
}

// h = y + 0.25*(nq+nk+v1+v2) ; v1/v2 are col block 1536 of the bn'd QKV bufs
__global__ void comb2_kernel(const float* __restrict__ y,
                             const float* __restrict__ nq, const float* __restrict__ nk,
                             const float* __restrict__ bnq1, const float* __restrict__ bnq2,
                             float* __restrict__ h, int n)
{
    int i = blockIdx.x * TB + threadIdx.x;
    if (i >= n) return;
    int r = i / 768, c = i - r * 768;
    float v1 = bnq1[(size_t)r * 2304 + 1536 + c];
    float v2 = bnq2[(size_t)r * 2304 + 1536 + c];
    h[i] = y[i] + 0.25f * (nq[i] + nk[i] + v1 + v2);
}

__global__ void add2_kernel(const float* __restrict__ a, int lda,
                            const float* __restrict__ b, int ldb,
                            float* __restrict__ o, int ldo, int rows, int cols)
{
    int i = blockIdx.x * TB + threadIdx.x;
    if (i >= rows * cols) return;
    int r = i / cols, c = i - r * cols;
    o[(size_t)r * ldo + c] = a[(size_t)r * lda + c] + b[(size_t)r * ldb + c];
}

// --------------------------- orchestration ---------------------------------

extern "C" void kernel_launch(void* const* d_in, const int* in_sizes, int n_in,
                              void* d_out, int out_size, void* d_ws, size_t ws_size,
                              hipStream_t stream)
{
    (void)in_sizes; (void)n_in; (void)out_size; (void)ws_size;

    const int NN = 4096, D = 512, H = 768, F = 3072, C3 = 2304;

    // Input indices: setup_inputs() insertion order, params flattened depth-
    // first in dict insertion order.
    // 0:x 1:x_noise2 2:A 3:Ad 4:gnn_q 5:gnn_k 6:gnn_v 7:norm_g 8:norm_b
    // 9..26: g1 (bn1_g,bn1_b,bn2_g,bn2_b,Wq,bq,Wk,bk,Wv,bv,Wo,bo,W1,b1,W2,b2,Wfc,bfc)
    // 27..44: g2 (same layout)  45:new_gnn_q 46:new_gnn_k
    // 47..62: nf (ln_g,ln_b,Wq,bq,Wk,bk,Wv,bv,Wo,bo,W1,b1,W2,b2,Wfc,bfc)
    auto IN = [&](int i) -> const float* { return (const float*)d_in[i]; };

    // ---- workspace carve ----
    char* ws = (char*)d_ws;
    size_t cur = 0;
    auto alloc = [&](size_t bytes) -> void* {
        void* p = ws + cur;
        cur += (bytes + 255) & ~(size_t)255;
        return p;
    };
    bf16_t* Ab     = (bf16_t*)alloc((size_t)NN * NN * 2);
    bf16_t* Adb    = (bf16_t*)alloc((size_t)NN * NN * 2);
    bf16_t* xb     = (bf16_t*)alloc((size_t)NN * D * 2);
    bf16_t* xn2b   = (bf16_t*)alloc((size_t)NN * D * 2);
    bf16_t* wqkv_t = (bf16_t*)alloc((size_t)C3 * D * 2);
    bf16_t* XWt    = (bf16_t*)alloc((size_t)C3 * NN * 2);
    bf16_t* bfNH   = (bf16_t*)alloc((size_t)NN * H * 2);
    bf16_t* bfNH2  = (bf16_t*)alloc((size_t)NN * H * 2);
    bf16_t* o1b    = (bf16_t*)alloc((size_t)NN * H * 2);   // out1 (bf16)
    bf16_t* o2b    = (bf16_t*)alloc((size_t)NN * H * 2);   // out2 (bf16)
    bf16_t* f1b    = (bf16_t*)alloc((size_t)NN * F * 2);
    bf16_t* ttb    = (bf16_t*)alloc((size_t)H * NN * 2);
    bf16_t* wslot  = (bf16_t*)alloc((size_t)F * H * 2);
    float* bnQKV1  = (float*)alloc((size_t)NN * C3 * 4);
    float* bnQKV2  = (float*)alloc((size_t)NN * C3 * 4);
    float* nq      = (float*)alloc((size_t)NN * H * 4);
    float* nk      = (float*)alloc((size_t)NN * H * 4);
    float* XW      = (float*)alloc((size_t)NN * C3 * 4);   // reused as raw QKV
    float* QKVn    = (float*)alloc((size_t)NN * C3 * 4);
    float* t1      = (float*)alloc((size_t)NN * H * 4);
    float* yb      = (float*)alloc((size_t)NN * H * 4);
    float* hbuf    = (float*)alloc((size_t)NN * H * 4);
    float* hb      = (float*)alloc((size_t)NN * H * 4);
    float* f2      = (float*)alloc((size_t)NN * H * 4);

    auto nb = [](long long n) { return (unsigned)((n + TB - 1) / TB); };
    auto gemm = [&](const bf16_t* A, const bf16_t* Bt, float* C, bf16_t* Cbf,
                    const float* bias, float bsc, int act, int M, int N, int K) {
        gemm_bf16_kernel<<<dim3(N / 128, M / 128), TB, 0, stream>>>(
            A, Bt, C, Cbf, bias, bsc, act, M, N, K);
    };
    auto cvt = [&](const float* s, int ld, bf16_t* d, int r, int c) {
        cvt_kernel<<<nb((long long)r * c), TB, 0, stream>>>(s, ld, d, r, c);
    };
    auto tconv = [&](const float* s, bf16_t* d, int R, int C) {
        tconv_kernel<<<dim3(C / 32, R / 32), dim3(32, 8), 0, stream>>>(s, d, R, C);
    };
    auto bn = [&](const float* z, int ldz, float* o, int ldo, const float* g,
                  const float* b, int cols, int gmod) {
        bn_kernel<<<cols / 32, dim3(32, 8), 0, stream>>>(z, ldz, o, ldo, g, b, NN, gmod);
    };

    // ---- precision staging: convert the big fp32 operands to bf16 once ----
    cvt(IN(0), D, xb,   NN, D);
    cvt(IN(1), D, xn2b, NN, D);
    cvt(IN(2), NN, Ab,  NN, NN);
    cvt(IN(3), NN, Adb, NN, NN);
    tconv(IN(4), wqkv_t,                    D, H);  // gnn_q -> rows 0..767 of [C3,D]
    tconv(IN(5), wqkv_t + (size_t)H * D,     D, H); // gnn_k
    tconv(IN(6), wqkv_t + (size_t)2 * H * D, D, H); // gnn_v

    // ---- stage A: bnQKV = BN(Adj @ (x @ [Wq|Wk|Wv]))  (x2) ----
    auto stageA = [&](const bf16_t* xv, const bf16_t* Adj, float* bnout) {
        gemm(xv, wqkv_t, XW, nullptr, nullptr, 0.f, 0, NN, C3, D);  // x @ W
        tconv(XW, XWt, NN, C3);                                     // -> [2304,4096] bf16
        gemm(Adj, XWt, XW, nullptr, nullptr, 0.f, 0, NN, C3, NN);   // Adj @ (xW)
        bn(XW, C3, bnout, C3, IN(7), IN(8), C3, H);                 // norm tiled x3
    };
    stageA(xb,   Ab,  bnQKV1);
    stageA(xn2b, Adb, bnQKV2);

    // ---- stage B: transformer block (softmax over len-1 kv is identity) ----
    auto runTF = [&](const float* qkv, int pb, bf16_t* outb) {
        bn(qkv, C3, QKVn, C3, IN(pb + 0), IN(pb + 1), C3, H);          // qn|kn|vn
        cvt(QKVn + 1536, C3, bfNH, NN, H);                             // vn -> bf16
        tconv(IN(pb + 8), wslot, H, H);                                // Wv^T
        gemm(bfNH, wslot, nullptr, bfNH2, IN(pb + 9), 1.f, 0, NN, H, H);  // vn@Wv+bv
        tconv(IN(pb + 10), wslot, H, H);                               // Wo^T
        gemm(bfNH2, wslot, yb, nullptr, IN(pb + 11), 1.f, 0, NN, H, H);   // y
        comb1_kernel<<<nb((long long)NN * H), TB, 0, stream>>>(QKVn, yb, hbuf, NN * H);
        bn(hbuf, H, hb, H, IN(pb + 2), IN(pb + 3), H, H);              // bn2
        cvt(hb, H, bfNH, NN, H);
        tconv(IN(pb + 12), wslot, H, F);                               // W1^T
        gemm(bfNH, wslot, nullptr, f1b, IN(pb + 13), 1.f, /*gelu*/1, NN, F, H);
        tconv(IN(pb + 14), wslot, F, H);                               // W2^T
        gemm(f1b, wslot, f2, nullptr, IN(pb + 15), 1.f, 0, NN, H, F);
        add2_kernel<<<nb((long long)NN * H), TB, 0, stream>>>(hbuf, H, f2, H, hbuf, H, NN, H);
        cvt(hbuf, H, bfNH, NN, H);
        tconv(IN(pb + 16), wslot, H, H);                               // Wfc^T
        gemm(bfNH, wslot, nullptr, outb, IN(pb + 17), 1.f, 0, NN, H, H);
    };
    runTF(bnQKV1, 9,  o1b);
    runTF(bnQKV2, 27, o2b);

    // ---- stage C: nq = A @ (out1 @ new_gnn_q), nk = Ad @ (out2 @ new_gnn_k)
    auto stageC = [&](const bf16_t* ob, const float* Wg, const bf16_t* Adj, float* nout) {
        tconv(Wg, wslot, H, H);
        gemm(ob, wslot, t1, nullptr, nullptr, 0.f, 0, NN, H, H);
        tconv(t1, ttb, NN, H);                                         // [768,4096] bf16
        gemm(Adj, ttb, nout, nullptr, nullptr, 0.f, 0, NN, H, NN);
    };
    stageC(o1b, IN(45), Ab,  nq);
    stageC(o2b, IN(46), Adb, nk);

    // ---- stage D: new_former ----
    // mha = ((v1+v2)@Wv + 2*bv) @ Wo + bo ; h = mha + (nq+nk+v1+v2)/4 ; + FFN(LN)
    add2_kernel<<<nb((long long)NN * H), TB, 0, stream>>>(
        bnQKV1 + 1536, C3, bnQKV2 + 1536, C3, hb, H, NN, H);           // vs = v1+v2
    cvt(hb, H, bfNH, NN, H);
    tconv(IN(53), wslot, H, H);                                        // nf Wv^T
    gemm(bfNH, wslot, nullptr, bfNH2, IN(54), 2.0f, 0, NN, H, H);      // + 2*bv
    tconv(IN(55), wslot, H, H);                                        // nf Wo^T
    gemm(bfNH2, wslot, yb, nullptr, IN(56), 1.f, 0, NN, H, H);
    comb2_kernel<<<nb((long long)NN * H), TB, 0, stream>>>(yb, nq, nk, bnQKV1, bnQKV2, hbuf, NN * H);
    ln_kernel<<<NN, TB, 0, stream>>>(hbuf, hb, IN(47), IN(48), H);
    cvt(hb, H, bfNH, NN, H);
    tconv(IN(57), wslot, H, F);                                        // W1^T
    gemm(bfNH, wslot, nullptr, f1b, IN(58), 1.f, /*gelu*/1, NN, F, H);
    tconv(IN(59), wslot, F, H);                                        // W2^T
    gemm(f1b, wslot, f2, nullptr, IN(60), 1.f, 0, NN, H, F);
    add2_kernel<<<nb((long long)NN * H), TB, 0, stream>>>(hbuf, H, f2, H, hbuf, H, NN, H);
    cvt(hbuf, H, bfNH, NN, H);
    tconv(IN(61), wslot, H, H);                                        // Wfc^T
    gemm(bfNH, wslot, (float*)d_out, nullptr, IN(62), 1.f, 0, NN, H, H);
}
